// SparseLinear_33079838114394
// MI455X (gfx1250) — matrix operand order repaired and checked by reading, not compiled, not run on hardware
//
#include <hip/hip_runtime.h>

#define MDIM 3072
#define KDIM 768

typedef __attribute__((ext_vector_type(16))) __bf16 v16bf;
typedef __attribute__((ext_vector_type(8)))  float  v8f;
typedef __attribute__((ext_vector_type(4)))  float  f32x4;

union BFV {
  unsigned short u[16];
  uint4 q[2];
  v16bf v;
};
static_assert(sizeof(BFV) == 32, "BFV must be 32 bytes");

// float -> bf16 round-to-nearest-even (finite inputs)
__device__ __forceinline__ unsigned short f2bf_rne(float f) {
  unsigned u = __builtin_bit_cast(unsigned, f);
  unsigned r = (u + 0x7FFFu + ((u >> 16) & 1u)) >> 16;
  return (unsigned short)r;
}
__device__ __forceinline__ float bf2f(unsigned short h) {
  unsigned u = ((unsigned)h) << 16;
  return __builtin_bit_cast(float, u);
}

// ---------------------------------------------------------------------------
// Kernel 1: zero the dense bf16 weight workspace (Wh | Wl)
// ---------------------------------------------------------------------------
__global__ void zero_kernel(uint4* __restrict__ p, int n16) {
  int i = blockIdx.x * blockDim.x + threadIdx.x;
  int stride = gridDim.x * blockDim.x;
  uint4 z; z.x = 0u; z.y = 0u; z.z = 0u; z.w = 0u;
  for (; i < n16; i += stride) p[i] = z;
}

// ---------------------------------------------------------------------------
// Kernel 2: CSR -> dense [M,K] bf16 hi + lo residual. One block per row.
// ---------------------------------------------------------------------------
__global__ void densify_kernel(const float* __restrict__ values,
                               const int*   __restrict__ row_offsets,
                               const int*   __restrict__ column_indices,
                               unsigned short* __restrict__ Wh,
                               unsigned short* __restrict__ Wl) {
  const int r = blockIdx.x;
  const int s = row_offsets[r];
  const int e = row_offsets[r + 1];
  for (int j = s + (int)threadIdx.x; j < e; j += (int)blockDim.x) {
    const int   c = column_indices[j];
    const float v = values[j];
    const unsigned short h = f2bf_rne(v);
    const unsigned short l = f2bf_rne(v - bf2f(h));
    Wh[(size_t)r * KDIM + c] = h;
    Wl[(size_t)r * KDIM + c] = l;
  }
}

// ---------------------------------------------------------------------------
// Kernel 2b: split x (fp32) into bf16 hi / lo arrays, 4 elements per thread.
// ---------------------------------------------------------------------------
__global__ void splitx_kernel(const float4* __restrict__ x4,
                              ushort4* __restrict__ xh4,
                              ushort4* __restrict__ xl4, int n4) {
  int i = blockIdx.x * blockDim.x + threadIdx.x;
  if (i >= n4) return;
  const float4 f = x4[i];
  ushort4 h, l;
  h.x = f2bf_rne(f.x); l.x = f2bf_rne(f.x - bf2f(h.x));
  h.y = f2bf_rne(f.y); l.y = f2bf_rne(f.y - bf2f(h.y));
  h.z = f2bf_rne(f.z); l.z = f2bf_rne(f.z - bf2f(h.z));
  h.w = f2bf_rne(f.w); l.w = f2bf_rne(f.w - bf2f(h.w));
  xh4[i] = h;
  xl4[i] = l;
}

// ---------------------------------------------------------------------------
// Kernel 3 (fast path): pure-WMMA GEMM, A/B already bf16 in memory.
// Wave: 16(n) x 64(m); block: 8 waves = 128(n) x 64(m).
// ---------------------------------------------------------------------------
__global__ __launch_bounds__(256) void spmm_wmma_pre_kernel(
    const unsigned short* __restrict__ xh,
    const unsigned short* __restrict__ xl,
    const unsigned short* __restrict__ Wh,
    const unsigned short* __restrict__ Wl,
    const float*          __restrict__ bias,
    float*                __restrict__ out) {
  const int lane = threadIdx.x & 31;
  const int wave = threadIdx.x >> 5;
  const int m0   = blockIdx.x * 64;
  const int n0   = blockIdx.y * 128 + wave * 16;
  const int half = lane >> 4;
  const int lid  = lane & 15;

  const size_t arow = (size_t)(n0 + lid) * KDIM;

  v8f acc[4] = {};

  for (int k0 = 0; k0 < KDIM; k0 += 32) {
    // A layout (16x32 bf16): lane<16 -> K k0+0..7 & k0+16..23,
    //                        lane>=16 -> K k0+8..15 & k0+24..31.
    const size_t aofs = arow + (size_t)(k0 + half * 8);
    BFV ah, al;
    ah.q[0] = *(const uint4*)(xh + aofs);
    ah.q[1] = *(const uint4*)(xh + aofs + 16);
    al.q[0] = *(const uint4*)(xl + aofs);
    al.q[1] = *(const uint4*)(xl + aofs + 16);

#pragma unroll
    for (int t = 0; t < 4; ++t) {
      const size_t wofs = (size_t)(m0 + t * 16 + lid) * KDIM + (size_t)(k0 + half * 8);
      BFV bhu, blu;
      bhu.q[0] = *(const uint4*)(Wh + wofs);
      bhu.q[1] = *(const uint4*)(Wh + wofs + 16);
      blu.q[0] = *(const uint4*)(Wl + wofs);
      blu.q[1] = *(const uint4*)(Wl + wofs + 16);
      const v16bf bh = bhu.v;
      const v16bf bl = blu.v;

      acc[t] = __builtin_amdgcn_wmma_f32_16x16x32_bf16(
          false, ah.v, false, bh, (short)0, acc[t], false, false);
      acc[t] = __builtin_amdgcn_wmma_f32_16x16x32_bf16(
          false, ah.v, false, bl, (short)0, acc[t], false, false);
      acc[t] = __builtin_amdgcn_wmma_f32_16x16x32_bf16(
          false, al.v, false, bh, (short)0, acc[t], false, false);
    }
  }

#pragma unroll
  for (int t = 0; t < 4; ++t) {
    const int   m  = m0 + t * 16 + lid;
    const float bv = bias[m];
    float* orow = out + (size_t)(n0 + half * 8) * MDIM + m;
#pragma unroll
    for (int r = 0; r < 8; ++r) {
      orow[(size_t)r * MDIM] = acc[t][r] + bv;
    }
  }
}

// ---------------------------------------------------------------------------
// Kernel 3 (fallback path): split x on the fly (small workspace).
// ---------------------------------------------------------------------------
__global__ __launch_bounds__(256) void spmm_wmma_fly_kernel(
    const float*          __restrict__ x,
    const unsigned short* __restrict__ Wh,
    const unsigned short* __restrict__ Wl,
    const float*          __restrict__ bias,
    float*                __restrict__ out) {
  const int lane = threadIdx.x & 31;
  const int wave = threadIdx.x >> 5;
  const int m0   = blockIdx.x * 64;
  const int n0   = blockIdx.y * 128 + wave * 16;
  const int half = lane >> 4;
  const int lid  = lane & 15;

  const float* xrow = x + (size_t)(n0 + lid) * KDIM;

  v8f acc[4] = {};

  for (int k0 = 0; k0 < KDIM; k0 += 32) {
    const float* p1 = xrow + k0 + half * 8;
    const f32x4 a0 = *(const f32x4*)(p1);
    const f32x4 a1 = *(const f32x4*)(p1 + 4);
    const f32x4 a2 = *(const f32x4*)(p1 + 16);
    const f32x4 a3 = *(const f32x4*)(p1 + 20);

    float fv[16];
#pragma unroll
    for (int e = 0; e < 4; ++e) {
      fv[e]      = a0[e];
      fv[4 + e]  = a1[e];
      fv[8 + e]  = a2[e];
      fv[12 + e] = a3[e];
    }

    BFV ah, al;
#pragma unroll
    for (int e = 0; e < 16; ++e) {
      const unsigned short h = f2bf_rne(fv[e]);
      ah.u[e] = h;
      al.u[e] = f2bf_rne(fv[e] - bf2f(h));
    }

#pragma unroll
    for (int t = 0; t < 4; ++t) {
      const size_t wofs = (size_t)(m0 + t * 16 + lid) * KDIM + (size_t)(k0 + half * 8);
      BFV bhu, blu;
      bhu.q[0] = *(const uint4*)(Wh + wofs);
      bhu.q[1] = *(const uint4*)(Wh + wofs + 16);
      blu.q[0] = *(const uint4*)(Wl + wofs);
      blu.q[1] = *(const uint4*)(Wl + wofs + 16);
      const v16bf bh = bhu.v;
      const v16bf bl = blu.v;

      acc[t] = __builtin_amdgcn_wmma_f32_16x16x32_bf16(
          false, ah.v, false, bh, (short)0, acc[t], false, false);
      acc[t] = __builtin_amdgcn_wmma_f32_16x16x32_bf16(
          false, ah.v, false, bl, (short)0, acc[t], false, false);
      acc[t] = __builtin_amdgcn_wmma_f32_16x16x32_bf16(
          false, al.v, false, bh, (short)0, acc[t], false, false);
    }
  }

#pragma unroll
  for (int t = 0; t < 4; ++t) {
    const int   m  = m0 + t * 16 + lid;
    const float bv = bias[m];
    float* orow = out + (size_t)(n0 + half * 8) * MDIM + m;
#pragma unroll
    for (int r = 0; r < 8; ++r) {
      orow[(size_t)r * MDIM] = acc[t][r] + bv;
    }
  }
}

// ---------------------------------------------------------------------------
extern "C" void kernel_launch(void* const* d_in, const int* in_sizes, int n_in,
                              void* d_out, int out_size, void* d_ws, size_t ws_size,
                              hipStream_t stream) {
  const float* values         = (const float*)d_in[0];
  const float* bias           = (const float*)d_in[1];
  const float* x              = (const float*)d_in[2];
  // d_in[3] = row_indices (identity; unused)
  const int*   row_offsets    = (const int*)d_in[4];
  const int*   column_indices = (const int*)d_in[5];
  float* out = (float*)d_out;

  const int ntok = in_sizes[2] / KDIM;  // B*S = 16384

  unsigned short* Wh = (unsigned short*)d_ws;
  unsigned short* Wl = Wh + (size_t)MDIM * KDIM;

  // 1) zero Wh|Wl (4*M*K bytes -> M*K/4 uint4)
  const int n16 = (MDIM * KDIM) / 4;
  zero_kernel<<<512, 256, 0, stream>>>((uint4*)d_ws, n16);

  // 2) scatter CSR into dense split-bf16 weight
  densify_kernel<<<MDIM, 256, 0, stream>>>(values, row_offsets, column_indices, Wh, Wl);

  const size_t wbytes  = (size_t)4 * MDIM * KDIM;                 // Wh+Wl
  const size_t xbytes  = (size_t)4 * ntok * KDIM;                 // xh+xl (bf16 each)
  dim3 grid(MDIM / 64, ntok / 128);

  if (ws_size >= wbytes + xbytes) {
    // fast path: precompute x hi/lo split once, GEMM is pure loads + WMMA
    unsigned short* xh = Wl + (size_t)MDIM * KDIM;
    unsigned short* xl = xh + (size_t)ntok * KDIM;

    const int n4 = (ntok * KDIM) / 4;
    splitx_kernel<<<(n4 + 255) / 256, 256, 0, stream>>>(
        (const float4*)x, (ushort4*)xh, (ushort4*)xl, n4);

    spmm_wmma_pre_kernel<<<grid, 256, 0, stream>>>(xh, xl, Wh, Wl, bias, out);
  } else {
    // fallback: split x on the fly inside the GEMM
    spmm_wmma_fly_kernel<<<grid, 256, 0, stream>>>(x, Wh, Wl, bias, out);
  }
}
